// HEAT_v2_68753836474686
// MI455X (gfx1250) — matrix operand order repaired and checked by reading, not compiled, not run on hardware
//
#include <hip/hip_runtime.h>

#define NB   200000
#define NG   50000
#define NN   250000
#define E0C  500000
#define E1C  50000
#define E2C  50000
#define EE   600000
#define H    64
#define ETE  16
#define EAE  16
#define OUTC 3
#define LAYERS 3
#define NEG  0.2f

typedef __attribute__((ext_vector_type(16))) _Float16 v16h;
typedef __attribute__((ext_vector_type(8)))  _Float16 v8h;
typedef __attribute__((ext_vector_type(8)))  float    v8f;

__device__ __forceinline__ float leaky(float x) { return x >= 0.f ? x : NEG * x; }

// order-preserving float -> uint key (for atomicMax-based segment max)
__device__ __forceinline__ unsigned f2key(float f) {
    unsigned b = __float_as_uint(f);
    return b ^ (unsigned)(((int)b >> 31) | 0x80000000);
}
__device__ __forceinline__ float key2f(unsigned k) {
    unsigned b = (k & 0x80000000u) ? (k ^ 0x80000000u) : ~k;
    return __uint_as_float(b);
}

// ---- WMMA fragment builders (ISA 7.12.2 16-bit layouts, wave32) ----
// A (M=16 rows x lda halves, row-major): lane holds row M=lane&15;
// halves are two contiguous 8-element runs -> 128-bit loads (ds or global).
__device__ __forceinline__ v16h load_a(const _Float16* As, int lda, int lane, int kbase) {
    int half = lane >> 4, m = lane & 15;
    const _Float16* p = As + m * lda + kbase + 8 * half;
    v8h lo = *(const v8h*)(p);
    v8h hi = *(const v8h*)(p + 16);
    return __builtin_shufflevector(lo, hi, 0, 1, 2, 3, 4, 5, 6, 7, 8, 9, 10, 11, 12, 13, 14, 15);
}
// B stored transposed in LDS: Bt[n*ldk + k]. Lane (col n=lane&15) reads two
// 8-half runs: k = kbase + 8*half + (0..7) and kbase + 16 + 8*half + (0..7).
__device__ __forceinline__ v16h load_bt(const _Float16* Bt, int ldk, int lane, int kbase, int nbase) {
    int n = nbase + (lane & 15);
    const _Float16* p = Bt + n * ldk + kbase + 8 * (lane >> 4);
    v8h lo = *(const v8h*)(p);
    v8h hi = *(const v8h*)(p + 16);
    return __builtin_shufflevector(lo, hi, 0, 1, 2, 3, 4, 5, 6, 7, 8, 9, 10, 11, 12, 13, 14, 15);
}

// ---- K1: per-type input embed + relu -> x[N,H] ----
__global__ void k_embed_nodes(const float* __restrict__ xb, const float* __restrict__ xg,
                              const float* __restrict__ Wb, const float* __restrict__ bb,
                              const float* __restrict__ Wg, const float* __restrict__ bg,
                              float* __restrict__ x) {
    long idx = blockIdx.x * (long)blockDim.x + threadIdx.x;
    if (idx >= (long)NN * H) return;
    int node = (int)(idx / H), c = (int)(idx % H);
    float s;
    if (node < NB) {
        s = bb[c];
        const float* r = xb + (long)node * 16;
#pragma unroll
        for (int k = 0; k < 16; ++k) s += r[k] * Wb[k * H + c];
    } else {
        s = bg[c];
        const float* r = xg + (long)(node - NB) * 8;
#pragma unroll
        for (int k = 0; k < 8; ++k) s += r[k] * Wg[k * H + c];
    }
    x[idx] = s > 0.f ? s : 0.f;
}

// ---- K2: per-edge-type attr embed + relu -> ea16[E,H] (f16) ----
__global__ void k_embed_edges(const float* __restrict__ ea0, const float* __restrict__ ea1,
                              const float* __restrict__ ea2, const float* __restrict__ We,
                              const float* __restrict__ be, _Float16* __restrict__ ea16) {
    long idx = blockIdx.x * (long)blockDim.x + threadIdx.x;
    if (idx >= (long)EE * H) return;
    int e = (int)(idx / H), c = (int)(idx % H);
    int t, r;
    if (e < E0C)            { t = 0; r = e; }
    else if (e < E0C + E1C) { t = 1; r = e - E0C; }
    else                    { t = 2; r = e - E0C - E1C; }
    const float* ap = (t == 0) ? ea0 : (t == 1) ? ea1 : ea2;
    float s = be[t * H + c];
#pragma unroll
    for (int k = 0; k < 9; ++k) s += ap[(long)r * 9 + k] * We[(t * 9 + k) * H + c];
    ea16[idx] = (_Float16)(s > 0.f ? s : 0.f);
}

// ---- K3: HeteroLinear via WMMA: h = x @ Wn[type] + bn[type] ----
// 64 nodes per block (4 row-tiles), both weight types staged transposed in LDS;
// B-fragments loaded once per wave, reused across the 4 row tiles.
__global__ void __launch_bounds__(128) k_hetero(const float* __restrict__ x,
                                                const float* __restrict__ Wn,
                                                const float* __restrict__ bn,
                                                float* __restrict__ h) {
    __shared__ __align__(16) _Float16 Bt[2 * H * 72];  // [type][n][k], padded rows
    __shared__ __align__(16) _Float16 As[64 * 72];     // 64 node rows x 64 (+pad)
    int tid = threadIdx.x;
    int nrow0 = blockIdx.x * 64;
    for (int i = tid; i < 2 * H * H; i += 128) {
        int t = i >> 12, rem = i & 4095;
        int k = rem >> 6, n = rem & 63;
        Bt[t * (H * 72) + n * 72 + k] = (_Float16)Wn[t * (H * H) + k * H + n];
    }
    for (int i = tid; i < 64 * H; i += 128) {
        int r = i >> 6, c = i & 63;
        int grow = nrow0 + r;
        As[r * 72 + c] = (_Float16)(grow < NN ? x[(long)grow * H + c] : 0.f);
    }
    __syncthreads();
    int wave = tid >> 5, lane = tid & 31;
    int nb = wave * 16;
    int t = (nrow0 < NB) ? 0 : 1;  // NB % 64 == 0: type is block-uniform
    const _Float16* Btt = Bt + t * (H * 72);
    v16h b0 = load_bt(Btt, 72, lane, 0, nb);
    v16h b1 = load_bt(Btt, 72, lane, 32, nb);
    int half = lane >> 4, col = nb + (lane & 15);
    float bias = bn[t * H + col];
#pragma unroll
    for (int tt = 0; tt < 4; ++tt) {
        int row0 = nrow0 + tt * 16;
        if (row0 >= NN) break;
        const _Float16* At = As + tt * 16 * 72;
        v16h a0 = load_a(At, 72, lane, 0);
        v16h a1 = load_a(At, 72, lane, 32);
        v8f acc = {};
        acc = __builtin_amdgcn_wmma_f32_16x16x32_f16(false, a0, false, b0, (short)0, acc, false, false);
        acc = __builtin_amdgcn_wmma_f32_16x16x32_f16(false, a1, false, b1, (short)0, acc, false, false);
#pragma unroll
        for (int v = 0; v < 8; ++v) {
            int m = v + 8 * half;
            h[(long)(row0 + m) * H + col] = acc[v] + bias;
        }
    }
}

// ---- K4: zero accumulator + softmax state ----
__global__ void k_init(float* __restrict__ x, unsigned* __restrict__ mkey, float* __restrict__ ssum) {
    long idx = blockIdx.x * (long)blockDim.x + threadIdx.x;
    if (idx < (long)NN * H) x[idx] = 0.f;
    if (idx < NN) { mkey[idx] = 0u; ssum[idx] = 0.f; }
}

// ---- K5: eae = leaky(ea @ Wea) via WMMA; A-fragments straight from global ----
__global__ void __launch_bounds__(128) k_eae(const _Float16* __restrict__ ea16,
                                             const float* __restrict__ Wea,
                                             float* __restrict__ eae) {
    __shared__ __align__(16) _Float16 Bt[EAE * 72];  // Wea^T: [n][k], padded
    int tid = threadIdx.x;
    for (int i = tid; i < H * EAE; i += 128) {
        int k = i >> 4, n = i & 15;
        Bt[n * 72 + k] = (_Float16)Wea[k * EAE + n];
    }
    __syncthreads();
    int wave = tid >> 5, lane = tid & 31;
    int e0 = blockIdx.x * 64 + wave * 16;  // one 16-edge tile per wave
    v16h b0 = load_bt(Bt, 72, lane, 0, 0);
    v16h b1 = load_bt(Bt, 72, lane, 32, 0);
    const _Float16* At = ea16 + (long)e0 * H;  // row-major 16x64, global b128 loads
    v16h a0 = load_a(At, H, lane, 0);
    v16h a1 = load_a(At, H, lane, 32);
    v8f acc = {};
    acc = __builtin_amdgcn_wmma_f32_16x16x32_f16(false, a0, false, b0, (short)0, acc, false, false);
    acc = __builtin_amdgcn_wmma_f32_16x16x32_f16(false, a1, false, b1, (short)0, acc, false, false);
    int half = lane >> 4, n = lane & 15;
#pragma unroll
    for (int v = 0; v < 8; ++v) {
        int m = v + 8 * half;
        eae[(long)(e0 + m) * EAE + n] = leaky(acc[v]);
    }
}

// ---- K6: attention logits (wave per edge) + segment max via uint atomicMax ----
__global__ void k_logits(const float* __restrict__ h, const float* __restrict__ eae,
                         const float* __restrict__ Etl, const float* __restrict__ Wa,
                         const int* __restrict__ src, const int* __restrict__ dst,
                         const int* __restrict__ ety, float* __restrict__ logit,
                         unsigned* __restrict__ mkey) {
    __shared__ float Was[2 * H + ETE + EAE];  // 160
    int tid = threadIdx.x;
    if (tid < 2 * H + ETE + EAE) Was[tid] = Wa[tid];
    __syncthreads();
    int e = blockIdx.x * 8 + (tid >> 5);
    if (e >= EE) return;
    int lane = tid & 31;
    int s = src[e], d = dst[e];
    float p = 0.f;
    p += h[(long)d * H + 2 * lane]     * Was[2 * lane];
    p += h[(long)d * H + 2 * lane + 1] * Was[2 * lane + 1];
    p += h[(long)s * H + 2 * lane]     * Was[H + 2 * lane];
    p += h[(long)s * H + 2 * lane + 1] * Was[H + 2 * lane + 1];
    if (lane < 16) {
        p += leaky(Etl[ety[e] * ETE + lane]) * Was[2 * H + lane];
    } else {
        int k = lane - 16;
        p += eae[(long)e * EAE + k] * Was[2 * H + ETE + k];
    }
#pragma unroll
    for (int off = 16; off > 0; off >>= 1) p += __shfl_xor(p, off);
    if (lane == 0) {
        float lg = leaky(p);
        logit[e] = lg;
        atomicMax(mkey + d, f2key(lg));
    }
}

// ---- K7: ex = exp(logit - m[dst]); segment sum ----
__global__ void k_expsum(float* __restrict__ logit, const int* __restrict__ dst,
                         const unsigned* __restrict__ mkey, float* __restrict__ ssum) {
    long e = blockIdx.x * (long)blockDim.x + threadIdx.x;
    if (e >= EE) return;
    int d = dst[e];
    float ex = expf(logit[e] - key2f(mkey[d]));
    logit[e] = ex;
    atomicAdd(ssum + d, ex);
}

// ---- K8: alpha = ex / (s[dst] + eps) ----
__global__ void k_alpha(const float* __restrict__ ex, const float* __restrict__ ssum,
                        const int* __restrict__ dst, float* __restrict__ alpha) {
    long e = blockIdx.x * (long)blockDim.x + threadIdx.x;
    if (e >= EE) return;
    alpha[e] = ex[e] / (ssum[dst[e]] + 1e-16f);
}

// ---- K9: message GEMM via WMMA (K=80 padded), 64 edges/block, scatter-add ----
__global__ void __launch_bounds__(128) k_message(const float* __restrict__ h,
                                                 const float* __restrict__ eae,
                                                 const float* __restrict__ alpha,
                                                 const int* __restrict__ src,
                                                 const int* __restrict__ dst,
                                                 const float* __restrict__ Wl,
                                                 const float* __restrict__ bl,
                                                 float* __restrict__ xacc) {
    __shared__ __align__(16) _Float16 Bt[H * 104];  // Wl^T: [n][k], rows 80..95 zero, pad 104
    __shared__ __align__(16) _Float16 As[64 * 104]; // 64 edges x 96 (+pad)
    __shared__ int   srcs[64];
    __shared__ int   dsts[64];
    __shared__ float alph[64];
    int tid = threadIdx.x;
    int e0 = blockIdx.x * 64;
    if (tid < 64) { srcs[tid] = src[e0 + tid]; dsts[tid] = dst[e0 + tid]; alph[tid] = alpha[e0 + tid]; }
    for (int i = tid; i < 96 * H; i += 128) {
        int k = i >> 6, n = i & 63;
        Bt[n * 104 + k] = (_Float16)((k < H + EAE) ? Wl[k * H + n] : 0.f);
    }
    __syncthreads();
    for (int i = tid; i < 64 * 96; i += 128) {
        int r = i / 96, c = i % 96;
        float v;
        if (c < H)            v = h[(long)srcs[r] * H + c];           // gathered x_j
        else if (c < H + EAE) v = eae[(long)(e0 + r) * EAE + (c - H)];
        else                  v = 0.f;                                 // K padding
        As[r * 104 + c] = (_Float16)v;
    }
    __syncthreads();
    int wave = tid >> 5, lane = tid & 31;
    int nb = wave * 16;
    v16h b0 = load_bt(Bt, 104, lane, 0, nb);
    v16h b1 = load_bt(Bt, 104, lane, 32, nb);
    v16h b2 = load_bt(Bt, 104, lane, 64, nb);
    int half = lane >> 4, col = nb + (lane & 15);
    float bias = bl[col];
#pragma unroll
    for (int tt = 0; tt < 4; ++tt) {
        const _Float16* At = As + tt * 16 * 104;
        v16h a0 = load_a(At, 104, lane, 0);
        v16h a1 = load_a(At, 104, lane, 32);
        v16h a2 = load_a(At, 104, lane, 64);
        v8f acc = {};
        acc = __builtin_amdgcn_wmma_f32_16x16x32_f16(false, a0, false, b0, (short)0, acc, false, false);
        acc = __builtin_amdgcn_wmma_f32_16x16x32_f16(false, a1, false, b1, (short)0, acc, false, false);
        acc = __builtin_amdgcn_wmma_f32_16x16x32_f16(false, a2, false, b2, (short)0, acc, false, false);
#pragma unroll
        for (int v = 0; v < 8; ++v) {
            int m = tt * 16 + v + 8 * half;
            float val = (acc[v] + bias) * alph[m];
            atomicAdd(&xacc[(long)dsts[m] * H + col], val);
        }
    }
}

// ---- K10: relu in place ----
__global__ void k_relu(float* __restrict__ x) {
    long idx = blockIdx.x * (long)blockDim.x + threadIdx.x;
    if (idx < (long)NN * H) x[idx] = fmaxf(x[idx], 0.f);
}

// ---- K11: per-type output projection + sigmoid ----
__global__ void k_out(const float* __restrict__ x, const float* __restrict__ Wb,
                      const float* __restrict__ bb, const float* __restrict__ Wg,
                      const float* __restrict__ bg, float* __restrict__ out) {
    long idx = blockIdx.x * (long)blockDim.x + threadIdx.x;
    if (idx >= (long)NN * OUTC) return;
    int node = (int)(idx / OUTC), c = (int)(idx % OUTC);
    const float* W; float s;
    if (node < NB) { W = Wb; s = bb[c]; } else { W = Wg; s = bg[c]; }
    const float* r = x + (long)node * H;
#pragma unroll
    for (int k = 0; k < H; ++k) s += r[k] * W[k * OUTC + c];
    out[idx] = 1.f / (1.f + expf(-s));  // bus block then gen block == flat node*3+c
}

extern "C" void kernel_launch(void* const* d_in, const int* in_sizes, int n_in,
                              void* d_out, int out_size, void* d_ws, size_t ws_size,
                              hipStream_t stream) {
    const float* x_bus     = (const float*)d_in[0];
    const float* x_gen     = (const float*)d_in[1];
    const float* ea0       = (const float*)d_in[2];
    const float* ea1       = (const float*)d_in[3];
    const float* ea2       = (const float*)d_in[4];
    const float* W_lin_bus = (const float*)d_in[5];
    const float* b_lin_bus = (const float*)d_in[6];
    const float* W_lin_gen = (const float*)d_in[7];
    const float* b_lin_gen = (const float*)d_in[8];
    const float* W_elin    = (const float*)d_in[9];
    const float* b_elin    = (const float*)d_in[10];
    const float* Wn        = (const float*)d_in[11];
    const float* bn        = (const float*)d_in[12];
    const float* Et        = (const float*)d_in[13];
    const float* Wea       = (const float*)d_in[14];
    const float* Watt      = (const float*)d_in[15];
    const float* Wl        = (const float*)d_in[16];
    const float* bl        = (const float*)d_in[17];
    const float* W_out_bus = (const float*)d_in[18];
    const float* b_out_bus = (const float*)d_in[19];
    const float* W_out_gen = (const float*)d_in[20];
    const float* b_out_gen = (const float*)d_in[21];
    const int*   src       = (const int*)d_in[22];
    const int*   dst       = (const int*)d_in[23];
    const int*   ety       = (const int*)d_in[24];
    float* out = (float*)d_out;

    // workspace carve-up
    float*    x     = (float*)d_ws;                       // N*H
    float*    h     = x + (size_t)NN * H;                 // N*H
    _Float16* ea16  = (_Float16*)(h + (size_t)NN * H);    // E*H f16
    float*    eae   = (float*)(ea16 + (size_t)EE * H);    // E*16
    float*    exb   = eae + (size_t)EE * EAE;             // E (logits -> ex)
    float*    alpha = exb + EE;                           // E
    unsigned* mkey  = (unsigned*)(alpha + EE);            // N
    float*    ssum  = (float*)(mkey + NN);                // N

    const long NH = (long)NN * H;
    dim3 B256(256), B128(128);

    k_embed_nodes<<<(unsigned)((NH + 255) / 256), B256, 0, stream>>>(
        x_bus, x_gen, W_lin_bus, b_lin_bus, W_lin_gen, b_lin_gen, x);
    k_embed_edges<<<(unsigned)(((long)EE * H + 255) / 256), B256, 0, stream>>>(
        ea0, ea1, ea2, W_elin, b_elin, ea16);

    for (int l = 0; l < LAYERS; ++l) {
        const float* Wn_l   = Wn   + (size_t)l * 2 * H * H;
        const float* bn_l   = bn   + (size_t)l * 2 * H;
        const float* Et_l   = Et   + (size_t)l * 3 * ETE;
        const float* Wea_l  = Wea  + (size_t)l * H * EAE;
        const float* Watt_l = Watt + (size_t)l * (2 * H + ETE + EAE);
        const float* Wl_l   = Wl   + (size_t)l * (H + EAE) * H;
        const float* bl_l   = bl   + (size_t)l * H;

        k_hetero<<<(NN + 63) / 64, B128, 0, stream>>>(x, Wn_l, bn_l, h);
        k_init<<<(unsigned)((NH + 255) / 256), B256, 0, stream>>>(x, mkey, ssum);
        k_eae<<<EE / 64, B128, 0, stream>>>(ea16, Wea_l, eae);
        k_logits<<<EE / 8, B256, 0, stream>>>(h, eae, Et_l, Watt_l, src, dst, ety, exb, mkey);
        k_expsum<<<(EE + 255) / 256, B256, 0, stream>>>(exb, dst, mkey, ssum);
        k_alpha<<<(EE + 255) / 256, B256, 0, stream>>>(exb, ssum, dst, alpha);
        k_message<<<EE / 64, B128, 0, stream>>>(h, eae, alpha, src, dst, Wl_l, bl_l, x);
        k_relu<<<(unsigned)((NH + 255) / 256), B256, 0, stream>>>(x);
    }

    k_out<<<(unsigned)(((long)NN * OUTC + 255) / 256), B256, 0, stream>>>(
        x, W_out_bus, b_out_bus, W_out_gen, b_out_gen, out);
}
